// SPINN_AttnIdea1_6485400617075
// MI455X (gfx1250) — matrix operand order repaired and checked by reading, not compiled, not run on hardware
//
#include <hip/hip_runtime.h>

typedef __bf16 bf16;
typedef __attribute__((ext_vector_type(16))) __bf16 v16bf;
typedef __attribute__((ext_vector_type(8)))  __bf16 v8bf;
typedef __attribute__((ext_vector_type(8)))  float   v8f;
typedef __attribute__((ext_vector_type(4)))  int     v4i;

#define SLOPEC 0.01f
#define Bsz 16384
#define Dsz 17
#define Hsz 256
#define Esz 64

// workspace layout (bf16 elements)
#define OFF_WMT   0
#define SZ_WMT    (2*Dsz*Hsz*Hsz)            // [l][i][o][h] transposed Wm
#define OFF_WFT   (OFF_WMT + SZ_WMT)
#define SZ_WFT    (Dsz*Esz*Hsz)              // [i][e][h] transposed Wf
#define OFF_W4    (OFF_WFT + SZ_WFT)
#define SZ_W4     (4*Esz*Esz)                // Wq,Wk,Wv,Wo row-major [e][c]
#define OFF_TOK   (OFF_W4 + SZ_W4)
#define SZ_TOK    (Bsz*Dsz*Esz)              // tokens bf16 [B][D][E]

// LDS strides (bf16 elements): 264*2 = 528 B rows -> 16B aligned, bank offset 4/row
#define WSTR 264
#define HSTR 264
#define TSTR 72
#define OSTR 68

// k_subnet LDS offsets (bytes): panel + 3 rotating h-buffers + scratch + WfT = 304128 B
#define P_OFF   0
#define B0_OFF  135168
#define B1_OFF  168960
#define B2_OFF  202752
#define SC_OFF  236544
#define WF_OFF  270336
#define SMEM1   304128
#define SMEM2   232704

// ---- CDNA5 async global->LDS staging (guarded; falls back to vmem+ds copy) ----
#define AS1 __attribute__((address_space(1)))
#define AS3 __attribute__((address_space(3)))
#if defined(__has_builtin)
#if __has_builtin(__builtin_amdgcn_global_load_async_to_lds_b128)
#define HAVE_ASYNC 1
#endif
#endif
#ifdef HAVE_ASYNC
#if __has_builtin(__builtin_amdgcn_s_wait_asynccnt)
#define ASYNC_WAIT() __builtin_amdgcn_s_wait_asynccnt(0)
#else
#define ASYNC_WAIT() asm volatile("s_wait_asynccnt 0x0" ::: "memory")
#endif
#else
#define ASYNC_WAIT()
#endif

__device__ __forceinline__ float lrelu_f(float v) { return v >= 0.f ? v : SLOPEC * v; }

__device__ __forceinline__ v16bf frag16(const bf16* p0, const bf16* p1) {
  v8bf lo = *(const v8bf*)p0;
  v8bf hi = *(const v8bf*)p1;
  return __builtin_shufflevector(lo, hi, 0,1,2,3,4,5,6,7,8,9,10,11,12,13,14,15);
}

// copy rows x (chunks*8) bf16 (src stride sstr) into LDS rows of stride dstr, 16B chunks
__device__ __forceinline__ void stage_lds(const bf16* __restrict__ src, bf16* __restrict__ dst,
                                          int rows, int chunks, int sstr, int dstr, int tid) {
  for (int it = tid; it < rows * chunks; it += 256) {
    int n = it / chunks, c8 = (it % chunks) * 8;
#ifdef HAVE_ASYNC
    // (int4 AS1* gsrc, int4 AS3* ldst, imm offset, imm cpol); AS3 from low 32 bits of generic addr
    __builtin_amdgcn_global_load_async_to_lds_b128(
        (AS1 v4i*)(__UINTPTR_TYPE__)(src + n * sstr + c8),
        (AS3 v4i*)(unsigned int)(__UINTPTR_TYPE__)(dst + n * dstr + c8), 0, 0);
#else
    *(v8bf*)(dst + n * dstr + c8) = *(const v8bf*)(src + n * sstr + c8);
#endif
  }
}

// ---------------- weight conversion / transpose kernels ----------------
__global__ void __launch_bounds__(256) k_conv_wm(const float* __restrict__ Wm, bf16* __restrict__ WmT) {
  int idx = blockIdx.x * 256 + threadIdx.x;
  if (idx >= 2 * Dsz * Hsz * Hsz) return;
  int h = idx & 255, o = (idx >> 8) & 255, li = idx >> 16;        // dst [li][o][h]
  WmT[idx] = (bf16)Wm[((size_t)li * 256 + h) * 256 + o];          // src [li][h][o]
}

__global__ void __launch_bounds__(256) k_conv_wf(const float* __restrict__ Wf, bf16* __restrict__ WfT) {
  int idx = blockIdx.x * 256 + threadIdx.x;
  if (idx >= Dsz * Esz * Hsz) return;
  int h = idx & 255, e = (idx >> 8) & 63, i = idx >> 14;          // dst [i][e][h]
  WfT[idx] = (bf16)Wf[((size_t)i * 256 + h) * 64 + e];            // src [i][h][e]
}

__global__ void __launch_bounds__(256) k_conv_w4(const float* __restrict__ Wq, const float* __restrict__ Wk,
                                                 const float* __restrict__ Wv, const float* __restrict__ Wo,
                                                 bf16* __restrict__ dst) {
  int idx = blockIdx.x * 256 + threadIdx.x;
  if (idx >= 4 * 4096) return;
  int which = idx >> 12, r = idx & 4095;
  const float* s = which == 0 ? Wq : which == 1 ? Wk : which == 2 ? Wv : Wo;
  dst[idx] = (bf16)s[r];  // y = t @ W.T => WT[n][k] = W row-major as-is
}

// ---------------- fused per-subnet MLP kernel ----------------
// 64x256 @ 256x256 GEMM. Wave owns 2 M-tiles x 4 N-tiles: each B-fragment feeds
// two WMMAs; B is software double-buffered so next ds_load issues before the
// current WMMA pair consumes its fragment (avoids s_wait_dscnt 0 drains).
__device__ __forceinline__ void gemm_mid(const bf16* __restrict__ A, const bf16* __restrict__ W,
                                         const float* __restrict__ bias, bf16* __restrict__ C,
                                         int lm, int lh, int w) {
  const int mp = w & 1;                 // M-tile pair {2mp, 2mp+1}
  const int ng = w >> 1;                // N-group: nt = ng*4 .. ng*4+3
  const int mr0 = (2 * mp) * 16;
  const int mr1 = (2 * mp + 1) * 16;
  const bf16* arow0 = A + (mr0 + lm) * HSTR;
  const bf16* arow1 = A + (mr1 + lm) * HSTR;
  v16bf af0[8], af1[8];
#pragma unroll
  for (int kk = 0; kk < 8; ++kk) {
    int k0 = kk * 32 + lh * 8;          // A: lh=0 -> K {0..7,16..23}, lh=1 -> {8..15,24..31}
    af0[kk] = frag16(arow0 + k0, arow0 + k0 + 16);
    af1[kk] = frag16(arow1 + k0, arow1 + k0 + 16);
  }
  float bn[4];
#pragma unroll
  for (int j = 0; j < 4; ++j) bn[j] = bias[(ng * 4 + j) * 16 + lm];
  for (int j = 0; j < 4; ++j) {
    int n = (ng * 4 + j) * 16 + lm;
    const bf16* wrow = W + n * WSTR;    // B: lane n, lh=0 -> K {0..7,16..23}, lh=1 -> {8..15,24..31}
    v8f c0 = {}, c1 = {};
    v16bf bcur = frag16(wrow + lh * 8, wrow + lh * 8 + 16);
#pragma unroll
    for (int kk = 0; kk < 7; ++kk) {
      int kb = (kk + 1) * 32 + lh * 8;
      v16bf bnxt = frag16(wrow + kb, wrow + kb + 16);   // prefetch next K-step
      c0 = __builtin_amdgcn_wmma_f32_16x16x32_bf16(false, af0[kk], false, bcur, (short)0, c0, false, false);
      c1 = __builtin_amdgcn_wmma_f32_16x16x32_bf16(false, af1[kk], false, bcur, (short)0, c1, false, false);
      bcur = bnxt;
    }
    c0 = __builtin_amdgcn_wmma_f32_16x16x32_bf16(false, af0[7], false, bcur, (short)0, c0, false, false);
    c1 = __builtin_amdgcn_wmma_f32_16x16x32_bf16(false, af1[7], false, bcur, (short)0, c1, false, false);
#pragma unroll
    for (int r = 0; r < 8; ++r) {
      C[(mr0 + r + lh * 8) * HSTR + n] = (bf16)lrelu_f(c0[r] + bn[j]);
      C[(mr1 + r + lh * 8) * HSTR + n] = (bf16)lrelu_f(c1[r] + bn[j]);
    }
  }
}

// tokens = A(64x256) @ WfT + bf -> global bf16, row-guarded. Wave: 2 M-tiles x 1 N-tile.
__device__ __forceinline__ void gemm_wf(const bf16* __restrict__ A, const bf16* __restrict__ Wft,
                                        const float* __restrict__ bfv, bf16* __restrict__ tokws,
                                        int i, int row0, int lm, int lh, int w) {
  const int mp = w & 1;
  const int nt = w >> 1;                // 4 N-tiles
  const int mr0 = (2 * mp) * 16;
  const int mr1 = (2 * mp + 1) * 16;
  const bf16* arow0 = A + (mr0 + lm) * HSTR;
  const bf16* arow1 = A + (mr1 + lm) * HSTR;
  const int n = nt * 16 + lm;
  const bf16* wrow = Wft + n * WSTR;
  const float bn = bfv[i * Esz + n];
  v8f c0 = {}, c1 = {};
#pragma unroll
  for (int kk = 0; kk < 8; ++kk) {
    int k0 = kk * 32 + lh * 8;
    v16bf a0 = frag16(arow0 + k0, arow0 + k0 + 16);
    v16bf a1 = frag16(arow1 + k0, arow1 + k0 + 16);
    int kb = kk * 32 + lh * 8;
    v16bf bfr = frag16(wrow + kb, wrow + kb + 16);
    c0 = __builtin_amdgcn_wmma_f32_16x16x32_bf16(false, a0, false, bfr, (short)0, c0, false, false);
    c1 = __builtin_amdgcn_wmma_f32_16x16x32_bf16(false, a1, false, bfr, (short)0, c1, false, false);
  }
#pragma unroll
  for (int r = 0; r < 8; ++r) {
    int row = row0 + mr0 + r + lh * 8;
    if (row < Bsz)
      tokws[((size_t)row * Dsz + i) * Esz + n] = (bf16)(c0[r] + bn);
  }
#pragma unroll
  for (int r = 0; r < 8; ++r) {
    int row = row0 + mr1 + r + lh * 8;
    if (row < Bsz)
      tokws[((size_t)row * Dsz + i) * Esz + n] = (bf16)(c1[r] + bn);
  }
}

__global__ void __launch_bounds__(256) k_subnet(
    const float* __restrict__ x, const float* __restrict__ W1, const float* __restrict__ b1,
    const float* __restrict__ bm, const float* __restrict__ bfv,
    const bf16* __restrict__ WmT, const bf16* __restrict__ WfT, bf16* __restrict__ tokws) {
  extern __shared__ char smem[];
  bf16* panel = (bf16*)(smem + P_OFF);
  bf16* buf0  = (bf16*)(smem + B0_OFF);
  bf16* buf1  = (bf16*)(smem + B1_OFF);
  bf16* buf2  = (bf16*)(smem + B2_OFF);
  bf16* scr   = (bf16*)(smem + SC_OFF);
  bf16* wft   = (bf16*)(smem + WF_OFF);
  const int tid = threadIdx.x;
  const int lane = tid & 31;
  const int w = tid >> 5;
  const int lm = lane & 15, lh = lane >> 4;
  const int i = blockIdx.y;
  const int r0 = blockIdx.x * 192;           // 3 chunks of 64 rows per block
  const float* bm0 = bm + (0 * Dsz + i) * Hsz;
  const float* bm1 = bm + (1 * Dsz + i) * Hsz;

  // P0: async-stage Wm0 panel + WfT, overlap with first-layer outer product
  stage_lds(WmT + (size_t)(0 * Dsz + i) * Hsz * Hsz, panel, 256, 32, 256, WSTR, tid);
  stage_lds(WfT + (size_t)i * Esz * Hsz, wft, 64, 32, 256, WSTR, tid);
  for (int c = 0; c < 3; ++c) {
    bf16* dst = c == 0 ? buf0 : c == 1 ? buf1 : buf2;
    for (int it = tid; it < 64 * 32; it += 256) {
      int m = it >> 5, hb = (it & 31) * 8;
      int row = r0 + c * 64 + m;
      float xv = row < Bsz ? x[(size_t)row * Dsz + i] : 0.f;
#pragma unroll
      for (int d = 0; d < 8; ++d) {
        float v = xv * W1[i * Hsz + hb + d] + b1[i * Hsz + hb + d];
        dst[m * HSTR + hb + d] = (bf16)lrelu_f(v);
      }
    }
  }
  ASYNC_WAIT();
  __syncthreads();

  // middle layer 0, rotating buffers (h0 consumed as h1 is produced)
  gemm_mid(buf0, panel, bm0, scr,  lm, lh, w); __syncthreads();   // h1(c0) -> scr
  gemm_mid(buf1, panel, bm0, buf0, lm, lh, w); __syncthreads();   // h1(c1) -> buf0
  gemm_mid(buf2, panel, bm0, buf1, lm, lh, w); __syncthreads();   // h1(c2) -> buf1

  // swap panel to Wm1 (single swap per 192 rows)
  stage_lds(WmT + (size_t)(1 * Dsz + i) * Hsz * Hsz, panel, 256, 32, 256, WSTR, tid);
  ASYNC_WAIT();
  __syncthreads();

  // middle layer 1 interleaved with Wf token projection
  gemm_mid(scr, panel, bm1, buf2, lm, lh, w); __syncthreads();    // h2(c0) -> buf2
  gemm_wf(buf2, wft, bfv, tokws, i, r0 + 0, lm, lh, w);           // tokens c0
  gemm_mid(buf0, panel, bm1, scr, lm, lh, w); __syncthreads();    // h2(c1) -> scr
  gemm_wf(scr, wft, bfv, tokws, i, r0 + 64, lm, lh, w);           // tokens c1
  gemm_mid(buf1, panel, bm1, buf0, lm, lh, w); __syncthreads();   // h2(c2) -> buf0
  gemm_wf(buf0, wft, bfv, tokws, i, r0 + 128, lm, lh, w);         // tokens c2
}

// ---------------- fused attention + pool + head kernel ----------------
// 16 batch rows per block -> 272 token rows = exactly 17 WMMA M-tiles.
__global__ void __launch_bounds__(256) k_attn(
    const bf16* __restrict__ tokws, const bf16* __restrict__ w4,
    const float* __restrict__ bq, const float* __restrict__ bk,
    const float* __restrict__ bv, const float* __restrict__ bo,
    const float* __restrict__ Wfc, const float* __restrict__ bfc,
    float* __restrict__ out) {
  extern __shared__ char smem[];
  bf16* tok  = (bf16*)(smem + 0);          // 272 x TSTR (39168 B)
  bf16* wlds = (bf16*)(smem + 39168);      // 4 x 64 x TSTR (36864 B)
  bf16* qld  = (bf16*)(smem + 76032);      // 272 x TSTR
  bf16* kld  = (bf16*)(smem + 115200);
  bf16* vld  = (bf16*)(smem + 154368);
  bf16* obuf = (bf16*)(smem + 193536);     // 272 x TSTR, ends 232704
  float* op     = (float*)(smem + 76032);  // reuse q+k regions: 272 x OSTR f32
  float* pooled = (float*)(smem + 39168);  // reuse weight region: 16 x 64 f32

  const int tid = threadIdx.x, lane = tid & 31, w = tid >> 5;
  const int lm = lane & 15, lh = lane >> 4;

  stage_lds(tokws + (size_t)blockIdx.x * 272 * 64, tok, 272, 8, 64, TSTR, tid);
  stage_lds(w4, wlds, 4 * 64, 8, 64, TSTR, tid);
  ASYNC_WAIT();
  __syncthreads();

  // Q/K/V projections: 3 mats x (17 M-tiles x 4 N-tiles), K=64 -> 2 wmma per tile
  for (int t = w; t < 3 * 68; t += 8) {
    int which = t / 68, r = t % 68;
    int mt = r % 17, nt = r / 17;
    int m = mt * 16 + lm, n = nt * 16 + lm;
    const bf16* arow = tok + m * TSTR;
    const bf16* wrow = wlds + which * 64 * TSTR + n * TSTR;
    v8f c = {};
#pragma unroll
    for (int kk = 0; kk < 2; ++kk) {
      int k0 = kk * 32 + lh * 8;
      v16bf a = frag16(arow + k0, arow + k0 + 16);
      int kb = kk * 32 + lh * 8;
      v16bf b = frag16(wrow + kb, wrow + kb + 16);
      c = __builtin_amdgcn_wmma_f32_16x16x32_bf16(false, a, false, b, (short)0, c, false, false);
    }
    const float* bias = which == 0 ? bq : (which == 1 ? bk : bv);
    bf16* dst = which == 0 ? qld : (which == 1 ? kld : vld);
    float bn = bias[n];
#pragma unroll
    for (int rr = 0; rr < 8; ++rr)
      dst[(mt * 16 + rr + lh * 8) * TSTR + n] = (bf16)(c[rr] + bn);
  }
  __syncthreads();

  // per-(batch,head) 17x17 attention in VALU (tiny; WMMA would waste padded tiles)
  for (int j = 0; j < 8; ++j) {
    int p = w * 8 + j;
    int b = p >> 2, hh = p & 3;
    if (lane < 17) {
      int r0 = b * 17;
      const bf16* qrow = qld + (r0 + lane) * TSTR + hh * 16;
      float s[17];
      float mx = -3.4e38f;
#pragma unroll
      for (int kj = 0; kj < 17; ++kj) {
        const bf16* krow = kld + (r0 + kj) * TSTR + hh * 16;
        float acc = 0.f;
#pragma unroll
        for (int d = 0; d < 16; ++d) acc += (float)qrow[d] * (float)krow[d];
        s[kj] = acc * 0.25f;     // 1/sqrt(16)
        mx = fmaxf(mx, s[kj]);
      }
      float sum = 0.f;
#pragma unroll
      for (int kj = 0; kj < 17; ++kj) { s[kj] = __expf(s[kj] - mx); sum += s[kj]; }
      float inv = 1.f / sum;
#pragma unroll
      for (int d = 0; d < 16; ++d) {
        float acc = 0.f;
#pragma unroll
        for (int kj = 0; kj < 17; ++kj)
          acc += s[kj] * (float)vld[(r0 + kj) * TSTR + hh * 16 + d];
        obuf[(r0 + lane) * TSTR + hh * 16 + d] = (bf16)(acc * inv);
      }
    }
  }
  __syncthreads();

  // out projection: o' = obuf @ Wo^T + bo  -> f32 LDS
  for (int t = w; t < 68; t += 8) {
    int mt = t % 17, nt = t / 17;
    int m = mt * 16 + lm, n = nt * 16 + lm;
    const bf16* arow = obuf + m * TSTR;
    const bf16* wrow = wlds + 3 * 64 * TSTR + n * TSTR;
    v8f c = {};
#pragma unroll
    for (int kk = 0; kk < 2; ++kk) {
      int k0 = kk * 32 + lh * 8;
      v16bf a = frag16(arow + k0, arow + k0 + 16);
      int kb = kk * 32 + lh * 8;
      v16bf b = frag16(wrow + kb, wrow + kb + 16);
      c = __builtin_amdgcn_wmma_f32_16x16x32_bf16(false, a, false, b, (short)0, c, false, false);
    }
    float bn = bo[n];
#pragma unroll
    for (int rr = 0; rr < 8; ++rr)
      op[(mt * 16 + rr + lh * 8) * OSTR + n] = c[rr] + bn;
  }
  __syncthreads();

  // mean pool over D=17 tokens
  for (int it = tid; it < 16 * 64; it += 256) {
    int b = it >> 6, e = it & 63;
    float s = 0.f;
#pragma unroll
    for (int d = 0; d < 17; ++d) s += op[(b * 17 + d) * OSTR + e];
    pooled[it] = s * (1.f / 17.f);
  }
  __syncthreads();

  // final head: lrelu(pooled @ Wfc + bfc)
  if (tid < 16) {
    float acc = bfc[0];
#pragma unroll
    for (int e = 0; e < 64; ++e) acc += pooled[tid * 64 + e] * Wfc[e];
    out[blockIdx.x * 16 + tid] = lrelu_f(acc);
  }
}

// ---------------- launcher ----------------
extern "C" void kernel_launch(void* const* d_in, const int* in_sizes, int n_in,
                              void* d_out, int out_size, void* d_ws, size_t ws_size,
                              hipStream_t stream) {
  (void)in_sizes; (void)n_in; (void)out_size; (void)ws_size;
  const float* x   = (const float*)d_in[0];
  const float* W1  = (const float*)d_in[1];
  const float* b1  = (const float*)d_in[2];
  const float* Wm  = (const float*)d_in[3];
  const float* bm  = (const float*)d_in[4];
  const float* Wf  = (const float*)d_in[5];
  const float* bfv = (const float*)d_in[6];
  const float* Wq  = (const float*)d_in[7];
  const float* bq  = (const float*)d_in[8];
  const float* Wk  = (const float*)d_in[9];
  const float* bk  = (const float*)d_in[10];
  const float* Wv  = (const float*)d_in[11];
  const float* bv  = (const float*)d_in[12];
  const float* Wo  = (const float*)d_in[13];
  const float* bo  = (const float*)d_in[14];
  const float* Wfc = (const float*)d_in[15];
  const float* bfc = (const float*)d_in[16];
  float* out = (float*)d_out;

  bf16* wsb = (bf16*)d_ws;
  bf16* WmT   = wsb + OFF_WMT;
  bf16* WfT   = wsb + OFF_WFT;
  bf16* W4    = wsb + OFF_W4;
  bf16* tokws = wsb + OFF_TOK;

  (void)hipFuncSetAttribute((const void*)k_subnet, hipFuncAttributeMaxDynamicSharedMemorySize, SMEM1);
  (void)hipFuncSetAttribute((const void*)k_attn,   hipFuncAttributeMaxDynamicSharedMemorySize, SMEM2);

  k_conv_wm<<<(2 * Dsz * Hsz * Hsz + 255) / 256, 256, 0, stream>>>(Wm, WmT);
  k_conv_wf<<<(Dsz * Esz * Hsz + 255) / 256, 256, 0, stream>>>(Wf, WfT);
  k_conv_w4<<<(4 * 4096 + 255) / 256, 256, 0, stream>>>(Wq, Wk, Wv, Wo, W4);

  dim3 g1((Bsz + 191) / 192, Dsz);           // 86 x 17 blocks, 192 rows each (tail-guarded)
  k_subnet<<<g1, 256, SMEM1, stream>>>(x, W1, b1, bm, bfv, WmT, WfT, tokws);

  k_attn<<<Bsz / 16, 256, SMEM2, stream>>>(tokws, W4, bq, bk, bv, bo, Wfc, bfc, out);
}
